// GAT_RPE_L2L_Encoder_55722905698615
// MI455X (gfx1250) — matrix operand order repaired and checked by reading, not compiled, not run on hardware
//
#include <hip/hip_runtime.h>
#include <hip/hip_bf16.h>
#include <stdint.h>
#include <stddef.h>

// ---------------------------------------------------------------------------
// Types for CDNA5 WMMA
// ---------------------------------------------------------------------------
typedef __attribute__((__ext_vector_type__(16))) __bf16 v16bf;
typedef __attribute__((__ext_vector_type__(8)))  float  v8f;

#define LN_EPS 1e-5f
#define NT 8   // 8 x 16 = 128 output columns per wave (one column panel)

__device__ __forceinline__ unsigned short f2bf(float f) {
  unsigned u = __float_as_uint(f);
  u += 0x7FFFu + ((u >> 16) & 1u);           // round-to-nearest-even
  return (unsigned short)(u >> 16);
}
// monotonic float<->uint mapping for atomicMax on floats
__device__ __forceinline__ unsigned encf(float f) {
  unsigned u = __float_as_uint(f);
  return (u & 0x80000000u) ? ~u : (u | 0x80000000u);
}
__device__ __forceinline__ float decf(unsigned e) {
  unsigned u = (e & 0x80000000u) ? (e ^ 0x80000000u) : ~e;
  return __uint_as_float(u);
}

// ---------------------------------------------------------------------------
// Weight pre-pack: f32 [K x ldn] -> bf16 in WMMA B-fragment order.
// Packed layout (dwords): [kChunk][panel][ (half*NT + t)*16 + l15 ][ d ]
// so each lane's 8-dword B fragment is contiguous (2 x b128 loads),
// where within a 32-K chunk:  k = 2*d + 16*half (+pair bit), col = panel*128
// + t*16 + l15.  2048 dwords per (chunk, panel).
// ---------------------------------------------------------------------------
__global__ void k_pack_w(const float* __restrict__ W, unsigned short* __restrict__ Wp,
                         int K, int ldn) {
  int i = blockIdx.x * blockDim.x + threadIdx.x;
  if (i >= K * ldn) return;
  int k = i / ldn, c = i - k * ldn;
  int chunk = k >> 5, kr = k & 31;
  int half = kr >> 4, d = (kr & 15) >> 1, p = kr & 1;
  int nPanels = ldn >> 7;
  int panel = c >> 7, t = (c >> 4) & 7, l15 = c & 15;
  size_t dw = (size_t)(chunk * nPanels + panel) * 2048 +
              (size_t)(((half * NT + t) * 16 + l15) * 8 + d);
  Wp[dw * 2 + p] = f2bf(W[i]);
}

// ---------------------------------------------------------------------------
// bf16 WMMA GEMM:  out = epilogue( A[MxK]_bf16 @ W[K x ldn]_bf16(packed) )
// 128-thread block = 4 independent waves; each wave: 16-row strip x 128 cols
// (NT=8 accumulators).  Per chunk, ALL fragments (A: 2 x b128, B: 16 x b128)
// are loaded up-front into distinct registers, then 8 WMMAs issue — in-order
// VMEM returns let the scheduler use partial s_wait_loadcnt and overlap load
// latency with the matrix pipe across the whole chunk.
// Epilogues: +bias, +residual, LayerNorm (ldn==128, gridDim.y==1), ReLU,
// f32 and/or bf16 stores.
// ---------------------------------------------------------------------------
__global__ __launch_bounds__(128) void k_gemm_bf16(
    const unsigned short* __restrict__ A,   // M x K (bf16 bits, row-major)
    const unsigned short* __restrict__ Wp,  // packed bf16 weights
    const float* __restrict__ bias,         // [ldn] or null
    const float* __restrict__ res,          // M x ldn or null (added pre-LN)
    const float* __restrict__ lng,          // LN gamma or null (no LN)
    const float* __restrict__ lnb,          // LN beta
    float*          __restrict__ outF,      // M x ldn or null
    unsigned short* __restrict__ outB,      // M x ldn (bf16) or null
    int M, int K, int ldn, int relu)
{
  const int lane = threadIdx.x & 31;
  const int wave = threadIdx.x >> 5;
  const int half = lane >> 4;
  const int l15  = lane & 15;
  const int row0 = (blockIdx.x * 4 + wave) * 16;
  const int panel = blockIdx.y;
  const int nPanels = ldn >> 7;
  const int colBase = panel * 128;
  if (row0 >= M) return;

  v8f zero = {0.f,0.f,0.f,0.f,0.f,0.f,0.f,0.f};
  v8f acc[NT];
#pragma unroll
  for (int t = 0; t < NT; ++t) acc[t] = zero;

  const unsigned short* aRow = A + (size_t)(row0 + l15) * K;
  const unsigned* wpd0 = (const unsigned*)Wp;
  const int fragBase = ((half * NT) * 16 + l15) * 8;   // lane's t=0 fragment dword

  union frag { v16bf v; uint4 q[2]; };

  const int nChunks = K >> 5;
  for (int ch = 0; ch < nChunks; ++ch) {
    const int k0 = ch << 5;
    const unsigned* wpd = wpd0 + (size_t)(ch * nPanels + panel) * 2048;

    if (ch + 1 < nChunks) {                 // gfx1250 global_prefetch_b8
      __builtin_prefetch((const void*)(aRow + k0 + 32), 0, 1);
      __builtin_prefetch((const void*)(wpd0 + (size_t)((ch + 1) * nPanels + panel) * 2048), 0, 1);
    }

    // ---- issue ALL loads for this chunk first (18 x b128 in flight) ----
    frag af;
    af.q[0] = *(const uint4*)(aRow + k0 + 16 * half);      // K[16h .. 16h+8)
    af.q[1] = *(const uint4*)(aRow + k0 + 16 * half + 8);  // K[16h+8 .. +8)

    frag bfr[NT];
#pragma unroll
    for (int t = 0; t < NT; ++t) {
      const uint4* bp = (const uint4*)(wpd + fragBase + t * 128);
      bfr[t].q[0] = bp[0];
      bfr[t].q[1] = bp[1];
    }

    // ---- then the 8 matrix ops (waits can retire progressively) ----
#pragma unroll
    for (int t = 0; t < NT; ++t) {
      acc[t] = __builtin_amdgcn_wmma_f32_16x16x32_bf16(
          false, af.v, false, bfr[t].v, (short)0, acc[t], false, false);
    }
  }

  const bool doLN = (lng != nullptr);
#pragma unroll
  for (int r = 0; r < 8; ++r) {
    int R = row0 + 8 * half + r;           // D layout: VGPR r, rows split by half
    float rowv[NT];
#pragma unroll
    for (int t = 0; t < NT; ++t) {
      int c = colBase + t * 16 + l15;
      float v = acc[t][r];
      if (bias) v += bias[c];
      if (res)  v += res[(size_t)R * ldn + c];
      rowv[t] = v;
    }
    if (doLN) {   // full 128-wide row lives in this half-wave (16 lanes x 8 tiles)
      float s = 0.f, s2 = 0.f;
#pragma unroll
      for (int t = 0; t < NT; ++t) { s += rowv[t]; s2 += rowv[t] * rowv[t]; }
      for (int m = 1; m < 16; m <<= 1) {
        s  += __shfl_xor(s,  m, 32);
        s2 += __shfl_xor(s2, m, 32);
      }
      float mu  = s * (1.f / 128.f);
      float var = s2 * (1.f / 128.f) - mu * mu;
      float rs  = rsqrtf(var + LN_EPS);
#pragma unroll
      for (int t = 0; t < NT; ++t) {
        int c = t * 16 + l15;
        rowv[t] = (rowv[t] - mu) * rs * lng[c] + lnb[c];
      }
    }
#pragma unroll
    for (int t = 0; t < NT; ++t) {
      float o = rowv[t];
      if (relu) o = fmaxf(o, 0.f);
      size_t idx = (size_t)R * ldn + colBase + t * 16 + l15;
      if (outF) outF[idx] = o;
      if (outB) outB[idx] = f2bf(o);
    }
  }
}

// ---------------------------------------------------------------------------
// RPE embed: ea[e,0:128] = relu(LN(edge_attrs[e,0:8] @ W + b))   (K=8, VALU)
// ---------------------------------------------------------------------------
__global__ __launch_bounds__(128) void k_rpe(
    const float* __restrict__ ea8, const float* __restrict__ w,
    const float* __restrict__ bias, const float* __restrict__ g,
    const float* __restrict__ bb, float* __restrict__ out, int E_)
{
  int e = blockIdx.x * 4 + (threadIdx.x >> 5);
  int lane = threadIdx.x & 31;
  if (e >= E_) return;
  float a[8];
#pragma unroll
  for (int i = 0; i < 8; ++i) a[i] = ea8[(size_t)e * 8 + i];
  float v[4]; float s = 0.f, s2 = 0.f;
#pragma unroll
  for (int j = 0; j < 4; ++j) {
    int c = lane + 32 * j;
    float acc = bias[c];
#pragma unroll
    for (int i = 0; i < 8; ++i) acc += a[i] * w[i * 128 + c];
    v[j] = acc; s += acc; s2 += acc * acc;
  }
  for (int m = 1; m < 32; m <<= 1) { s += __shfl_xor(s, m, 32); s2 += __shfl_xor(s2, m, 32); }
  float mu = s * (1.f / 128.f), var = s2 * (1.f / 128.f) - mu * mu;
  float rs = rsqrtf(var + LN_EPS);
#pragma unroll
  for (int j = 0; j < 4; ++j) {
    int c = lane + 32 * j;
    out[(size_t)e * 128 + c] = fmaxf((v[j] - mu) * rs * g[c] + bb[c], 0.f);
  }
}

// out = maybe_relu(LN(in + res)) per 128-wide row; one wave per row
__global__ __launch_bounds__(256) void k_ln_rows(
    const float* __restrict__ in, const float* __restrict__ res,
    float* __restrict__ out, const float* __restrict__ g,
    const float* __restrict__ b, int rows, int relu)
{
  int row = blockIdx.x * 8 + (threadIdx.x >> 5);
  int lane = threadIdx.x & 31;
  if (row >= rows) return;
  float v[4]; float s = 0.f, s2 = 0.f;
#pragma unroll
  for (int j = 0; j < 4; ++j) {
    int c = lane + 32 * j;
    float x = in[(size_t)row * 128 + c];
    if (res) x += res[(size_t)row * 128 + c];
    v[j] = x; s += x; s2 += x * x;
  }
  for (int m = 1; m < 32; m <<= 1) { s += __shfl_xor(s, m, 32); s2 += __shfl_xor(s2, m, 32); }
  float mu = s * (1.f / 128.f), var = s2 * (1.f / 128.f) - mu * mu;
  float rs = rsqrtf(var + LN_EPS);
#pragma unroll
  for (int j = 0; j < 4; ++j) {
    int c = lane + 32 * j;
    float o = (v[j] - mu) * rs * g[c] + b[c];
    if (relu) o = fmaxf(o, 0.f);
    out[(size_t)row * 128 + c] = o;
  }
}

// cat[e] = bf16([x[tgt], x[src], ea[e]])   (one block of 128 per edge)
__global__ __launch_bounds__(128) void k_gather_cat(
    const float* __restrict__ x, const float* __restrict__ ea,
    const long long* __restrict__ srcI, const long long* __restrict__ tgtI,
    unsigned short* __restrict__ cat)
{
  size_t e = blockIdx.x; int j = threadIdx.x;
  long long s = srcI[e], t = tgtI[e];
  unsigned short* ce = cat + e * 384;
  ce[j]       = f2bf(x[(size_t)t * 128 + j]);
  ce[128 + j] = f2bf(x[(size_t)s * 128 + j]);
  ce[256 + j] = f2bf(ea[e * 128 + j]);
}

// per-(edge,head) attention logit + segment max
__global__ void k_logits(const float* __restrict__ Q, const float* __restrict__ Km,
                         const long long* __restrict__ tgtI,
                         float* __restrict__ logits, unsigned* __restrict__ mEnc,
                         long long EH)
{
  long long i = (long long)blockIdx.x * blockDim.x + threadIdx.x;
  if (i >= EH) return;
  long long e = i >> 3; int h = (int)(i & 7);
  long long t = tgtI[e];
  const float* q = Q  + (size_t)t * 128 + h * 16;
  const float* k = Km + (size_t)e * 128 + h * 16;
  float s = 0.f;
#pragma unroll
  for (int d = 0; d < 16; ++d) s += q[d] * k[d];
  s *= 0.25f;                       // 1/sqrt(16)
  logits[i] = s;
  atomicMax(&mEnc[(size_t)t * 8 + h], encf(s));
}

// ex = exp(logit - segmax); segment-sum denominators
__global__ void k_expden(float* __restrict__ att, const unsigned* __restrict__ mEnc,
                         const long long* __restrict__ tgtI, float* __restrict__ dN,
                         long long EH)
{
  long long i = (long long)blockIdx.x * blockDim.x + threadIdx.x;
  if (i >= EH) return;
  long long e = i >> 3; int h = (int)(i & 7);
  long long t = tgtI[e];
  float ex = __expf(att[i] - decf(mEnc[(size_t)t * 8 + h]));
  att[i] = ex;
  atomicAdd(&dN[(size_t)t * 8 + h], ex);
}

// aggr[tgt] += attn * v   (o-projection deferred: linear, commutes with sum)
__global__ __launch_bounds__(128) void k_aggr(
    const float* __restrict__ att, const float* __restrict__ dN,
    const long long* __restrict__ tgtI, const float* __restrict__ V,
    float* __restrict__ aggr)
{
  size_t e = blockIdx.x; int j = threadIdx.x; int h = j >> 4;
  long long t = tgtI[e];
  float a = att[e * 8 + h] / (dN[(size_t)t * 8 + h] + 1e-16f);
  atomicAdd(&aggr[(size_t)t * 128 + j], a * V[e * 128 + j]);
}

__global__ void k_f32_to_bf16(const float* __restrict__ s,
                              unsigned short* __restrict__ d, size_t n) {
  size_t i = (size_t)blockIdx.x * blockDim.x + threadIdx.x;
  if (i < n) d[i] = f2bf(s[i]);
}
__global__ void k_fill_f32(float* p, size_t n, float v) {
  size_t i = (size_t)blockIdx.x * blockDim.x + threadIdx.x;
  if (i < n) p[i] = v;
}
__global__ void k_fill_enc(unsigned* p, size_t n, float v) {
  size_t i = (size_t)blockIdx.x * blockDim.x + threadIdx.x;
  if (i < n) p[i] = encf(v);
}

// ---------------------------------------------------------------------------
static inline int cdiv_i(long long a, long long b) { return (int)((a + b - 1) / b); }

extern "C" void kernel_launch(void* const* d_in, const int* in_sizes, int n_in,
                              void* d_out, int out_size, void* d_ws, size_t ws_size,
                              hipStream_t stream) {
  const int D = 128;
  const int N = in_sizes[0] / D;
  const int E = in_sizes[1] / 2;
  if (n_in < 7 + 3 * 22) return;

  const float*     lane_feats = (const float*)d_in[0];
  const long long* eidx       = (const long long*)d_in[1];
  const long long* srcI = eidx;
  const long long* tgtI = eidx + E;
  const float*     eattr = (const float*)d_in[2];
  const float *rpe_w = (const float*)d_in[3], *rpe_b = (const float*)d_in[4];
  const float *rpe_g = (const float*)d_in[5], *rpe_bb = (const float*)d_in[6];
  // per-layer leaves, insertion order of _layer_params
  enum { MEM_W, MEM_B, MEMLN_G, MEMLN_B, Q_W, K_W, V_W, O_W,
         EU_W, EU_B, EULN_G, EULN_B, ENLN_G, ENLN_B,
         F1_W, F1_B, F2_W, F2_B, N1_G, N1_B, N2_G, N2_B };
  const float* L[3][22];
  for (int l = 0; l < 3; ++l)
    for (int k = 0; k < 22; ++k) L[l][k] = (const float*)d_in[7 + l * 22 + k];

  // ---- workspace partition (256B aligned) --------------------------------
  char* base = (char*)d_ws; size_t off = 0;
  auto alloc = [&](size_t bytes) -> void* {
    void* p = base + off; off = (off + bytes + 255) & ~(size_t)255; return p;
  };
  float*          ea     = (float*)alloc((size_t)E * 128 * 4);
  unsigned short* cat    = (unsigned short*)alloc((size_t)E * 384 * 2);
  unsigned short* memBf  = (unsigned short*)alloc((size_t)E * 128 * 2);
  float*          kf     = (float*)alloc((size_t)E * 128 * 4);
  float*          vf     = (float*)alloc((size_t)E * 128 * 4);
  float*          tmpE   = (float*)alloc((size_t)E * 128 * 4);
  float*          att    = (float*)alloc((size_t)E * 8 * 4);
  unsigned short* xbf    = (unsigned short*)alloc((size_t)N * 128 * 2);
  float*          Qf     = (float*)alloc((size_t)N * 128 * 4);
  unsigned*       mN     = (unsigned*)alloc((size_t)N * 8 * 4);
  float*          dN     = (float*)alloc((size_t)N * 8 * 4);
  float*          aggr   = (float*)alloc((size_t)N * 128 * 4);
  unsigned short* aggrBf = (unsigned short*)alloc((size_t)N * 128 * 2);
  unsigned short* h1bf   = (unsigned short*)alloc((size_t)N * 256 * 2);
  unsigned short* wMem = (unsigned short*)alloc(384 * 128 * 2);
  unsigned short* wQ   = (unsigned short*)alloc(128 * 128 * 2);
  unsigned short* wK   = (unsigned short*)alloc(128 * 128 * 2);
  unsigned short* wV   = (unsigned short*)alloc(128 * 128 * 2);
  unsigned short* wO   = (unsigned short*)alloc(128 * 128 * 2);
  unsigned short* wEU  = (unsigned short*)alloc(128 * 128 * 2);
  unsigned short* wF1  = (unsigned short*)alloc(128 * 256 * 2);
  unsigned short* wF2  = (unsigned short*)alloc(256 * 128 * 2);
  (void)ws_size;

  auto cvt = [&](const float* s, unsigned short* d, size_t n) {
    k_f32_to_bf16<<<cdiv_i((long long)n, 256), 256, 0, stream>>>(s, d, n);
  };
  auto packw = [&](const float* s, unsigned short* d, int K, int ldn) {
    k_pack_w<<<cdiv_i((long long)K * ldn, 256), 256, 0, stream>>>(s, d, K, ldn);
  };
  auto gemm = [&](const unsigned short* A, const unsigned short* Wp,
                  const float* bias, const float* res,
                  const float* g, const float* b,
                  float* oF, unsigned short* oB,
                  int M, int K, int ldn, int relu) {
    dim3 gr((unsigned)cdiv_i(M, 64), (unsigned)(ldn / 128));
    k_gemm_bf16<<<gr, 128, 0, stream>>>(A, Wp, bias, res, g, b, oF, oB,
                                        M, K, ldn, relu);
  };

  float* x = (float*)d_out;                       // running node features
  hipMemcpyAsync(x, lane_feats, (size_t)N * 128 * 4,
                 hipMemcpyDeviceToDevice, stream);

  // ea = relu(LN(edge_attrs @ rpeW + b))
  k_rpe<<<cdiv_i(E, 4), 128, 0, stream>>>(eattr, rpe_w, rpe_b, rpe_g, rpe_bb, ea, E);

  const long long EH = (long long)E * 8;
  for (int l = 0; l < 3; ++l) {
    const float* const* P = L[l];
    packw(P[MEM_W], wMem, 384, 128);
    packw(P[Q_W],  wQ,  128, 128);
    packw(P[K_W],  wK,  128, 128);
    packw(P[V_W],  wV,  128, 128);
    packw(P[O_W],  wO,  128, 128);
    packw(P[EU_W], wEU, 128, 128);
    packw(P[F1_W], wF1, 128, 256);
    packw(P[F2_W], wF2, 256, 128);

    cvt(x, xbf, (size_t)N * 128);
    // Q per node (== per-edge q gathered by tgt)
    gemm(xbf, wQ, nullptr, nullptr, nullptr, nullptr, Qf, nullptr, N, 128, 128, 0);
    // cat = [x_tgt, x_src, ea] as bf16
    k_gather_cat<<<E, 128, 0, stream>>>(x, ea, srcI, tgtI, cat);
    // mem = relu(LN(cat @ Wmem + b))  -> bf16 (fused LN epilogue)
    gemm(cat, wMem, P[MEM_B], nullptr, P[MEMLN_G], P[MEMLN_B],
         nullptr, memBf, E, 384, 128, 1);
    // k, v
    gemm(memBf, wK, nullptr, nullptr, nullptr, nullptr, kf, nullptr, E, 128, 128, 0);
    gemm(memBf, wV, nullptr, nullptr, nullptr, nullptr, vf, nullptr, E, 128, 128, 0);
    // delta = relu(LN(mem @ Weu + b)); ea = LN(ea + delta)
    gemm(memBf, wEU, P[EU_B], nullptr, P[EULN_G], P[EULN_B],
         tmpE, nullptr, E, 128, 128, 1);
    k_ln_rows<<<cdiv_i(E, 8), 256, 0, stream>>>(tmpE, ea, ea,
                                                P[ENLN_G], P[ENLN_B], E, 0);
    // segment softmax + weighted aggregation
    k_fill_enc<<<cdiv_i((long long)N * 8, 256), 256, 0, stream>>>(mN, (size_t)N * 8, -3.0e38f);
    k_fill_f32<<<cdiv_i((long long)N * 8, 256), 256, 0, stream>>>(dN, (size_t)N * 8, 0.f);
    k_fill_f32<<<cdiv_i((long long)N * 128, 256), 256, 0, stream>>>(aggr, (size_t)N * 128, 0.f);
    k_logits<<<cdiv_i(EH, 256), 256, 0, stream>>>(Qf, kf, tgtI, att, mN, EH);
    k_expden<<<cdiv_i(EH, 256), 256, 0, stream>>>(att, mN, tgtI, dN, EH);
    k_aggr<<<E, 128, 0, stream>>>(att, dN, tgtI, vf, aggr);
    // x = LN(x + aggr @ Wo)   (o-proj after sum; fused residual+LN; xbf for FFN)
    cvt(aggr, aggrBf, (size_t)N * 128);
    gemm(aggrBf, wO, nullptr, x, P[N1_G], P[N1_B], x, xbf, N, 128, 128, 0);
    // FFN: h1 = relu(x@W1+b1) -> bf16 ; x = LN(x + h1@W2 + b2)
    gemm(xbf, wF1, P[F1_B], nullptr, nullptr, nullptr, nullptr, h1bf, N, 128, 256, 1);
    gemm(h1bf, wF2, P[F2_B], x, P[N2_G], P[N2_B], x, nullptr, N, 256, 128, 0);
  }
  (void)out_size;
}
